// LocalPoolPointnet_52183852646836
// MI455X (gfx1250) — matrix operand order repaired and checked by reading, not compiled, not run on hardware
//
#include <hip/hip_runtime.h>

// ---------------------------------------------------------------------------
// LocalPoolPointnet for MI455X (gfx1250): bf16 WMMA GEMMs, wave32.
// Resblock kernels: block = 256 threads = 8 waves = 2 row-groups x 4 col-groups.
// Each wave: 32 rows x 32 cols via 4 accumulators; block tile = 64 x 128.
// Per k-tile: 2 A-fragments + 2 B-fragments -> 4 WMMAs (1 ds_load_b128/WMMA).
// ---------------------------------------------------------------------------

typedef __bf16 bf16_t;
typedef __attribute__((ext_vector_type(16))) __bf16 bf16x16;
typedef __attribute__((ext_vector_type(8)))  __bf16 bf16x8;
typedef __attribute__((ext_vector_type(8)))  float  v8f;

#define NPTS   400000      // b * n
#define NPER   100000      // n per batch
#define NBATCH 4
#define R2     16384       // 128*128 bins
#define HD     128
#define TWOH   256
#define MTR    64          // rows per block (resblock kernels)
#define MTF    32          // rows per block (final kernel)

// ---- workspace byte offsets (all multiples of 256) ----
#define IDX_OFF   0L                       // 400000 * 4        = 1,600,000
#define NET_OFF   1600000L                 // 400000 * 128 * 4  = 204,800,000
#define BINS_OFF  206400000L               // 4*16384*128*4     = 33,554,432 (reused as planeSum)
#define CNT_OFF   239954432L               // 4*16384*4         = 262,144
#define WF_OFF    240216576L               // packed bf16 weights (425,984 elts)

// packed-weight element offsets inside WF region
#define W0F_OFF   0          // 5 * 32768
#define WSF_OFF   163840     // 5 * 32768
#define W1F_OFF   327680     // 5 * 16384
#define WCF_OFF   409600     // 16384

// ---------------------------------------------------------------------------
// float <-> order-preserving uint (for atomic max pooling)
// ---------------------------------------------------------------------------
__device__ __forceinline__ unsigned fenc(float f) {
  unsigned u = __float_as_uint(f);
  return (u & 0x80000000u) ? ~u : (u | 0x80000000u);
}
__device__ __forceinline__ float fdec(unsigned u) {
  unsigned v = (u & 0x80000000u) ? (u & 0x7FFFFFFFu) : ~u;
  return __uint_as_float(v);
}

// ---------------------------------------------------------------------------
// Repack a row-major f32 weight (K x 128) into bf16 WMMA B-fragment order:
// dst[((kt*8 + nt)*32 + lane)*16 + j] = W[kt*32 + (lane>=16?8:0) + (j<8?j:j+8)][nt*16 + (lane&15)]
// ---------------------------------------------------------------------------
__global__ void k_repack(const float* __restrict__ src, bf16_t* __restrict__ dst, int K) {
  int t = blockIdx.x * blockDim.x + threadIdx.x;
  if (t >= K * 128) return;
  int j    = t & 15;
  int lane = (t >> 4) & 31;
  int nt   = (t >> 9) & 7;
  int kt   = t >> 12;
  int k = kt * 32 + ((lane >= 16) ? 8 : 0) + (j < 8 ? j : j + 8);
  int n = nt * 16 + (lane & 15);
  dst[t] = (bf16_t)src[k * 128 + n];
}

// ---------------------------------------------------------------------------
// Bin index per point
// ---------------------------------------------------------------------------
__global__ void k_index(const float* __restrict__ inp, int* __restrict__ idx) {
  int p = blockIdx.x * blockDim.x + threadIdx.x;
  if (p >= NPTS) return;
  float x = inp[(long)p * 3 + 0];
  float y = inp[(long)p * 3 + 1];
  int i = (int)(x * 128.0f); i = i < 0 ? 0 : (i > 127 ? 127 : i);
  int j = (int)(y * 128.0f); j = j < 0 ? 0 : (j > 127 ? 127 : j);
  idx[p] = i + 128 * j;
}

__global__ void k_fill_u32(unsigned* __restrict__ p, unsigned val, long nelem) {
  long t = (long)blockIdx.x * blockDim.x + threadIdx.x;
  if (t < nelem) p[t] = val;
}

// ---------------------------------------------------------------------------
// scatter-max of net into bins (ordered-uint atomicMax)
// ---------------------------------------------------------------------------
__global__ void k_scatter_max(const float* __restrict__ net, const int* __restrict__ idx,
                              unsigned* __restrict__ bins) {
  long t = (long)blockIdx.x * blockDim.x + threadIdx.x;
  if (t >= (long)NPTS * HD) return;
  long p = t >> 7;
  int  f = (int)(t & 127);
  int  bi = (int)(p / NPER);
  unsigned key = fenc(net[t]);
  atomicMax(&bins[((long)bi * R2 + idx[p]) * HD + f], key);
}

// ---------------------------------------------------------------------------
// Wave-level GEMM accumulates.
// A layout per ISA 7.12.2: lane m=lane&15, khalf=(lane<16)?0:8,
//   element j<8 -> K = kbase+khalf+j ; j>=8 -> K = kbase+khalf+16+(j-8)
// ---------------------------------------------------------------------------
#define WMMA_BF16(A, B, C) \
  __builtin_amdgcn_wmma_f32_16x16x32_bf16(false, (A), false, (B), (short)0, (C), false, false)

// 2 row-tiles x 2 col-tiles: 2 A-frag loads + 2 B-frag loads -> 4 WMMAs.
template <int KT>
__device__ __forceinline__ void gemm_acc4(v8f& c00, v8f& c01, v8f& c10, v8f& c11,
                                          const bf16_t* lds, int stride,
                                          const bf16_t* __restrict__ wfrag,
                                          int lane, int nt0) {
  const int m     = lane & 15;
  const int khalf = (lane < 16) ? 0 : 8;
  const bf16_t* row0 = lds + m * stride;
  const bf16_t* row1 = lds + (m + 16) * stride;
#pragma unroll
  for (int kt = 0; kt < KT; ++kt) {
    bf16x16 a0, a1;
    *(bf16x8*)&a0       = *(const bf16x8*)(row0 + kt * 32 + khalf);
    *((bf16x8*)&a0 + 1) = *(const bf16x8*)(row0 + kt * 32 + khalf + 16);
    *(bf16x8*)&a1       = *(const bf16x8*)(row1 + kt * 32 + khalf);
    *((bf16x8*)&a1 + 1) = *(const bf16x8*)(row1 + kt * 32 + khalf + 16);
    bf16x16 b0 = *(const bf16x16*)(wfrag + ((long)(kt * 8 + nt0) * 32 + lane) * 16);
    bf16x16 b1 = *(const bf16x16*)(wfrag + ((long)(kt * 8 + nt0 + 1) * 32 + lane) * 16);
    c00 = WMMA_BF16(a0, b0, c00);
    c01 = WMMA_BF16(a0, b1, c01);
    c10 = WMMA_BF16(a1, b0, c10);
    c11 = WMMA_BF16(a1, b1, c11);
  }
}

// 2 row-tiles x 1 col-tile (used by k_final)
template <int KT>
__device__ __forceinline__ void gemm_acc2(v8f& c0, v8f& c1, const bf16_t* lds,
                                          int stride, const bf16_t* __restrict__ wfrag,
                                          int lane, int w) {
  const int m     = lane & 15;
  const int khalf = (lane < 16) ? 0 : 8;
  const bf16_t* row0 = lds + m * stride;
  const bf16_t* row1 = lds + (m + 16) * stride;
#pragma unroll
  for (int kt = 0; kt < KT; ++kt) {
    bf16x16 a0, a1;
    *(bf16x8*)&a0       = *(const bf16x8*)(row0 + kt * 32 + khalf);
    *((bf16x8*)&a0 + 1) = *(const bf16x8*)(row0 + kt * 32 + khalf + 16);
    *(bf16x8*)&a1       = *(const bf16x8*)(row1 + kt * 32 + khalf);
    *((bf16x8*)&a1 + 1) = *(const bf16x8*)(row1 + kt * 32 + khalf + 16);
    bf16x16 b = *(const bf16x16*)(wfrag + ((long)(kt * 8 + w) * 32 + lane) * 16);
    c0 = WMMA_BF16(a0, b, c0);
    c1 = WMMA_BF16(a1, b, c1);
  }
}

// ---------------------------------------------------------------------------
// Fused ResnetBlockFC core. lds_x = raw x (MTR x 256 bf16), lds_b = relu(x)
// (same shape) which is REUSED as the activation buffer (MTR x 128 bf16)
// after GEMM1 (extra barrier makes the aliasing safe).
//   net1 = relu(x)@W0 + b0 ; act = relu(net1)
//   out  = x@Ws + act@W1 + b1 -> net (f32, in place per-row)
// ---------------------------------------------------------------------------
__device__ __forceinline__ void resblock_core(float* __restrict__ net, long rowbase,
                                              const bf16_t* lds_x, bf16_t* lds_b,
                                              const bf16_t* w0f, const bf16_t* w1f,
                                              const bf16_t* wsf,
                                              const float* __restrict__ b0v,
                                              const float* __restrict__ b1v) {
  const int tid  = threadIdx.x;
  const int lane = tid & 31;
  const int w    = tid >> 5;
  const int wg   = w & 3;        // column group: cols [32wg, 32wg+32)
  const int rg   = w >> 2;       // row group: rows [32rg, 32rg+32)
  const int nt0  = 2 * wg;
  const int n0   = 32 * wg + (lane & 15);
  const int n1   = n0 + 16;
  const int hi8  = (lane >= 16) ? 8 : 0;
  const int rbase = rg * 32;

  // --- GEMM1: relu(x) @ W0 ---
  v8f c00, c01, c10, c11;
#pragma unroll
  for (int r = 0; r < 8; ++r) { c00[r] = 0.0f; c01[r] = 0.0f; c10[r] = 0.0f; c11[r] = 0.0f; }
  gemm_acc4<8>(c00, c01, c10, c11, lds_b + rbase * TWOH, TWOH, w0f, lane, nt0);
  __syncthreads();               // all waves done reading relu(x) from lds_b

  // act = relu(net1) into lds_b (reused as MTR x 128)
  bf16_t* act = lds_b;
  const float bA = b0v[n0], bB = b0v[n1];
#pragma unroll
  for (int r = 0; r < 8; ++r) {
    int m = rbase + r + hi8;
    act[m * HD + n0]        = (bf16_t)fmaxf(c00[r] + bA, 0.0f);
    act[m * HD + n1]        = (bf16_t)fmaxf(c01[r] + bB, 0.0f);
    act[(m + 16) * HD + n0] = (bf16_t)fmaxf(c10[r] + bA, 0.0f);
    act[(m + 16) * HD + n1] = (bf16_t)fmaxf(c11[r] + bB, 0.0f);
  }
  __syncthreads();

  // --- GEMM2: act @ W1  +  x @ Ws ---
  v8f d00, d01, d10, d11;
#pragma unroll
  for (int r = 0; r < 8; ++r) { d00[r] = 0.0f; d01[r] = 0.0f; d10[r] = 0.0f; d11[r] = 0.0f; }
  gemm_acc4<4>(d00, d01, d10, d11, act + rbase * HD, HD, w1f, lane, nt0);
  gemm_acc4<8>(d00, d01, d10, d11, lds_x + rbase * TWOH, TWOH, wsf, lane, nt0);

  const float cA = b1v[n0], cB = b1v[n1];
#pragma unroll
  for (int r = 0; r < 8; ++r) {
    int m = rbase + r + hi8;
    net[(rowbase + m) * HD + n0]      = d00[r] + cA;
    net[(rowbase + m) * HD + n1]      = d01[r] + cB;
    net[(rowbase + m + 16) * HD + n0] = d10[r] + cA;
    net[(rowbase + m + 16) * HD + n1] = d11[r] + cB;
  }
}

// ---------------------------------------------------------------------------
// Block 0: x = inputs @ Wpos + bpos computed on the fly into LDS, then resblock.
// ---------------------------------------------------------------------------
__global__ __launch_bounds__(256) void k_block0(
    const float* __restrict__ inp, const float* __restrict__ Wpos,
    const float* __restrict__ bpos, const bf16_t* __restrict__ w0f,
    const bf16_t* __restrict__ w1f, const bf16_t* __restrict__ wsf,
    const float* __restrict__ b0, const float* __restrict__ b1,
    float* __restrict__ net) {
  __shared__ __align__(16) bf16_t lds_x[MTR * TWOH];   // raw x     (32 KB)
  __shared__ __align__(16) bf16_t lds_b[MTR * TWOH];   // relu(x) -> act (32 KB)
  const long rowbase = (long)blockIdx.x * MTR;
  const int  tid = threadIdx.x;

  const float wp0 = Wpos[0 * TWOH + tid];
  const float wp1 = Wpos[1 * TWOH + tid];
  const float wp2 = Wpos[2 * TWOH + tid];
  const float bp  = bpos[tid];
#pragma unroll 8
  for (int m = 0; m < MTR; ++m) {
    long p = rowbase + m;
    float v = inp[p * 3 + 0] * wp0 + inp[p * 3 + 1] * wp1 + inp[p * 3 + 2] * wp2 + bp;
    lds_x[m * TWOH + tid] = (bf16_t)v;
    lds_b[m * TWOH + tid] = (bf16_t)fmaxf(v, 0.0f);
  }
  __syncthreads();
  resblock_core(net, rowbase, lds_x, lds_b, w0f, w1f, wsf, b0, b1);
}

// ---------------------------------------------------------------------------
// Blocks 1..4: x = concat[net, pooled(bins)] staged into LDS, then resblock.
// ---------------------------------------------------------------------------
__global__ __launch_bounds__(256) void k_blocki(
    const int* __restrict__ idx, const unsigned* __restrict__ bins,
    const bf16_t* __restrict__ w0f, const bf16_t* __restrict__ w1f,
    const bf16_t* __restrict__ wsf, const float* __restrict__ b0,
    const float* __restrict__ b1, float* __restrict__ net) {
  __shared__ __align__(16) bf16_t lds_x[MTR * TWOH];
  __shared__ __align__(16) bf16_t lds_b[MTR * TWOH];
  const long rowbase = (long)blockIdx.x * MTR;
  const int  tid = threadIdx.x;   // doubles as column 0..255 (uniform per wave)

#pragma unroll 8
  for (int m = 0; m < MTR; ++m) {
    long p = rowbase + m;
    float v;
    if (tid < HD) {
      v = net[p * HD + tid];
    } else {
      int bi = (int)(p / NPER);
      int bn = idx[p];
      v = fdec(bins[((long)bi * R2 + bn) * HD + (tid - HD)]);
    }
    lds_x[m * TWOH + tid] = (bf16_t)v;
    lds_b[m * TWOH + tid] = (bf16_t)fmaxf(v, 0.0f);
  }
  __syncthreads();
  resblock_core(net, rowbase, lds_x, lds_b, w0f, w1f, wsf, b0, b1);
}

// ---------------------------------------------------------------------------
// Final: c = relu(net) @ Wc + bc ; scatter-add into plane sums + counts
// ---------------------------------------------------------------------------
__global__ __launch_bounds__(256) void k_final(
    const float* __restrict__ net, const int* __restrict__ idx,
    const bf16_t* __restrict__ wcf, const float* __restrict__ bc,
    float* __restrict__ planeSum, float* __restrict__ cnt) {
  __shared__ __align__(16) bf16_t lds_a[MTF * HD];
  const long rowbase = (long)blockIdx.x * MTF;
  const int  tid = threadIdx.x;

#pragma unroll
  for (int e = 0; e < 16; ++e) {
    int flat = e * 256 + tid;
    int m = flat >> 7;
    int col = flat & 127;
    float v = net[(rowbase + m) * HD + col];
    lds_a[m * HD + col] = (bf16_t)fmaxf(v, 0.0f);
  }
  __syncthreads();

  const int lane = tid & 31;
  const int w    = tid >> 5;
  const int hi8  = (lane >= 16) ? 8 : 0;
  v8f c0, c1;
#pragma unroll
  for (int r = 0; r < 8; ++r) { c0[r] = 0.0f; c1[r] = 0.0f; }
  gemm_acc2<4>(c0, c1, lds_a, HD, wcf, lane, w);

  const int n = w * 16 + (lane & 15);
  const float bias = bc[n];
#pragma unroll
  for (int r = 0; r < 8; ++r) {
    int  m = r + hi8;
    long p0 = rowbase + m;
    long p1 = rowbase + m + 16;
    int  bi0 = (int)(p0 / NPER);
    int  bi1 = (int)(p1 / NPER);
    atomicAdd(&planeSum[((long)bi0 * R2 + idx[p0]) * HD + n], c0[r] + bias);
    atomicAdd(&planeSum[((long)bi1 * R2 + idx[p1]) * HD + n], c1[r] + bias);
  }
  if (tid < MTF) {
    long p = rowbase + tid;
    int  bi = (int)(p / NPER);
    atomicAdd(&cnt[(long)bi * R2 + idx[p]], 1.0f);
  }
}

// ---------------------------------------------------------------------------
// Divide by counts and transpose (b, R2, F) -> (b, F, R, R)
// ---------------------------------------------------------------------------
__global__ void k_out(const float* __restrict__ planeSum, const float* __restrict__ cnt,
                      float* __restrict__ out) {
  long t = (long)blockIdx.x * blockDim.x + threadIdx.x;
  if (t >= (long)NBATCH * HD * R2) return;
  int  bin = (int)(t % R2);
  long r   = t / R2;
  int  f   = (int)(r % HD);
  int  bi  = (int)(r / HD);
  float cv = cnt[(long)bi * R2 + bin];
  out[t] = planeSum[((long)bi * R2 + bin) * HD + f] / fmaxf(cv, 1.0f);
}

// ---------------------------------------------------------------------------
extern "C" void kernel_launch(void* const* d_in, const int* in_sizes, int n_in,
                              void* d_out, int out_size, void* d_ws, size_t ws_size,
                              hipStream_t stream) {
  const float* inputs = (const float*)d_in[0];
  const float* Wpos   = (const float*)d_in[1];
  const float* bpos   = (const float*)d_in[2];
  const float* W0     = (const float*)d_in[3];
  const float* b0     = (const float*)d_in[4];
  const float* W1     = (const float*)d_in[5];
  const float* b1     = (const float*)d_in[6];
  const float* Ws     = (const float*)d_in[7];
  const float* Wc     = (const float*)d_in[8];
  const float* bc     = (const float*)d_in[9];

  char* ws = (char*)d_ws;
  int*      idx      = (int*)(ws + IDX_OFF);
  float*    net      = (float*)(ws + NET_OFF);
  unsigned* bins     = (unsigned*)(ws + BINS_OFF);
  float*    planeSum = (float*)(ws + BINS_OFF);   // reused after pooling done
  float*    cnt      = (float*)(ws + CNT_OFF);
  bf16_t*   wf       = (bf16_t*)(ws + WF_OFF);

  // 1) repack weights to bf16 WMMA fragments
  for (int i = 0; i < 5; ++i) {
    k_repack<<<128, 256, 0, stream>>>(W0 + (long)i * 32768, wf + W0F_OFF + (long)i * 32768, 256);
    k_repack<<<128, 256, 0, stream>>>(Ws + (long)i * 32768, wf + WSF_OFF + (long)i * 32768, 256);
    k_repack<<< 64, 256, 0, stream>>>(W1 + (long)i * 16384, wf + W1F_OFF + (long)i * 16384, 128);
  }
  k_repack<<<64, 256, 0, stream>>>(Wc, wf + WCF_OFF, 128);

  // 2) bin indices
  k_index<<<(NPTS + 255) / 256, 256, 0, stream>>>(inputs, idx);

  // 3) block 0 (pos-embedding fused)
  k_block0<<<NPTS / MTR, 256, 0, stream>>>(inputs, Wpos, bpos,
                                           wf + W0F_OFF, wf + W1F_OFF, wf + WSF_OFF,
                                           b0, b1, net);

  // 4) blocks 1..4: pool-max then fused resblock
  const long binsElems = (long)NBATCH * R2 * HD;          // 8,388,608
  const long smElems   = (long)NPTS * HD;                 // 51,200,000
  for (int i = 1; i < 5; ++i) {
    k_fill_u32<<<(unsigned)((binsElems + 255) / 256), 256, 0, stream>>>(bins, 0u, binsElems);
    k_scatter_max<<<(unsigned)((smElems + 255) / 256), 256, 0, stream>>>(net, idx, bins);
    k_blocki<<<NPTS / MTR, 256, 0, stream>>>(idx, bins,
                                             wf + W0F_OFF + (long)i * 32768,
                                             wf + W1F_OFF + (long)i * 16384,
                                             wf + WSF_OFF + (long)i * 32768,
                                             b0 + (long)i * HD, b1 + (long)i * HD, net);
  }

  // 5) final projection + scatter-mean
  k_fill_u32<<<(unsigned)((binsElems + 255) / 256), 256, 0, stream>>>((unsigned*)planeSum, 0u, binsElems);
  k_fill_u32<<<(unsigned)(((long)NBATCH * R2 + 255) / 256), 256, 0, stream>>>((unsigned*)cnt, 0u, (long)NBATCH * R2);
  k_final<<<NPTS / MTF, 256, 0, stream>>>(net, idx, wf + WCF_OFF, bc, planeSum, cnt);

  // 6) divide + layout to (b, F, R, R)
  const long outElems = (long)NBATCH * HD * R2;
  k_out<<<(unsigned)((outElems + 255) / 256), 256, 0, stream>>>(planeSum, cnt, (float*)d_out);
}
